// ContinousPrior_12584254177992
// MI455X (gfx1250) — matrix operand order repaired and checked by reading, not compiled, not run on hardware
//
#include <hip/hip_runtime.h>
#include <math.h>

typedef _Float16 half_t;
typedef __attribute__((ext_vector_type(16))) _Float16 v16h;
typedef __attribute__((ext_vector_type(8)))  _Float16 v8h;
typedef __attribute__((ext_vector_type(8)))  float    v8f;

#define B_SZ   32
#define L_SEQ  64
#define Z_DIM  32
#define D_MODEL 512
#define NHEAD  8
#define HDIM   64
#define NLAYER 8
#define K_CB   8192
#define DFF_   2048

// Register blocking: each wave computes a (16*TM_) x (16*TN_) output block.
#define TM_ 2
#define TN_ 4

// ---------------------------------------------------------------------------
// Register-blocked WMMA GEMM: C[M x N] = A(f16, MxK row-major) * Wt(f16, NxK
// row-major, i.e. KxN weight stored transposed) + bias. One wave per 32x64
// output block: 8 v_wmma per K-step of 32, 12 b128 loads (A reused across the
// 4 N-tiles, B reused across the 2 M-tiles).
// Requires: M % 32 == 0, N % 64 == 0, K % 32 == 0.
// ---------------------------------------------------------------------------
__global__ __launch_bounds__(32) void gemm_wmma_kernel(
    const half_t* __restrict__ A, const half_t* __restrict__ Wt,
    const float* __restrict__ bias, float* __restrict__ Cf,
    half_t* __restrict__ Ch, int M, int N, int K, int relu)
{
    const int lane = threadIdx.x;
    const int tn = blockIdx.x * (16 * TN_);
    const int tm = blockIdx.y * (16 * TM_);
    const int l15 = lane & 15;
    const int koffA = (lane >> 4) * 8;       // lanes 16-31 hold K+8 block (A frag)
    const int koffB = (lane >> 4) * 8;       // lanes 16-31 hold K+8 block (B frag)

    const half_t* __restrict__ Arow[TM_];
#pragma unroll
    for (int i = 0; i < TM_; ++i)
        Arow[i] = A + (size_t)(tm + i * 16 + l15) * K + koffA;
    const half_t* __restrict__ Brow[TN_];
#pragma unroll
    for (int j = 0; j < TN_; ++j)
        Brow[j] = Wt + (size_t)(tn + j * 16 + l15) * K + koffB;

    v8f c[TM_][TN_];
#pragma unroll
    for (int i = 0; i < TM_; ++i)
#pragma unroll
        for (int j = 0; j < TN_; ++j)
            c[i][j] = (v8f){};

    for (int k0 = 0; k0 < K; k0 += 32) {
        v16h a[TM_], b[TN_];
#pragma unroll
        for (int i = 0; i < TM_; ++i) {
            v8h lo = *(const v8h*)(Arow[i] + k0);        // K: koffA..+7
            v8h hi = *(const v8h*)(Arow[i] + k0 + 16);   // K: koffA+16..+23
#pragma unroll
            for (int e = 0; e < 8; ++e) { a[i][e] = lo[e]; a[i][e + 8] = hi[e]; }
        }
#pragma unroll
        for (int j = 0; j < TN_; ++j) {
            v8h lo = *(const v8h*)(Brow[j] + k0);        // K: koffB..+7
            v8h hi = *(const v8h*)(Brow[j] + k0 + 16);   // K: koffB+16..+23
#pragma unroll
            for (int e = 0; e < 8; ++e) { b[j][e] = lo[e]; b[j][e + 8] = hi[e]; }
        }
#pragma unroll
        for (int i = 0; i < TM_; ++i)
#pragma unroll
            for (int j = 0; j < TN_; ++j)
                c[i][j] = __builtin_amdgcn_wmma_f32_16x16x32_f16(
                    false, a[i], false, b[j], (short)0, c[i][j], false, false);
    }

#pragma unroll
    for (int i = 0; i < TM_; ++i) {
        const int r0 = tm + i * 16 + ((lane >> 4) << 3);  // lanes16-31 = rows M+8
#pragma unroll
        for (int j = 0; j < TN_; ++j) {
            const int ncol = tn + j * 16 + l15;
            const float bv = bias ? bias[ncol] : 0.0f;
#pragma unroll
            for (int r = 0; r < 8; ++r) {
                float v = c[i][j][r] + bv;
                if (relu) v = v > 0.0f ? v : 0.0f;
                size_t idx = (size_t)(r0 + r) * N + ncol;
                if (Cf) Cf[idx] = v;
                if (Ch) Ch[idx] = (half_t)v;
            }
        }
    }
}

// ---------------------------------------------------------------------------
// Weight convert: f32 (layers x K x Nin)[:, :, col0:col0+Nout] -> f16 (layers x Nout x K)
// ---------------------------------------------------------------------------
__global__ void wT_kernel(const float* __restrict__ in, half_t* __restrict__ out,
                          int layers, int K, int Nin, int col0, int Nout)
{
    size_t idx = (size_t)blockIdx.x * blockDim.x + threadIdx.x;
    size_t total = (size_t)layers * Nout * K;
    if (idx >= total) return;
    int k = (int)(idx % K);
    int r = (int)(idx / K);
    int n = r % Nout;
    int l = r / Nout;
    out[idx] = (half_t)in[((size_t)l * K + k) * Nin + col0 + n];
}

// ---------------------------------------------------------------------------
// Context embed: ctx[b,l,d] = x_full[b,l,:] @ W_in + b_in + PE[l,d]
// x_full[b,0]=0, x_full[b,l]=x[b,l-1]
// ---------------------------------------------------------------------------
__global__ void ctx_kernel(const float* __restrict__ x, const float* __restrict__ Win,
                           const float* __restrict__ bin, float* __restrict__ ctx)
{
    size_t idx = (size_t)blockIdx.x * blockDim.x + threadIdx.x;
    if (idx >= (size_t)B_SZ * L_SEQ * D_MODEL) return;
    int d = (int)(idx % D_MODEL);
    int r = (int)(idx / D_MODEL);
    int l = r % L_SEQ;
    int b = r / L_SEQ;
    float acc = bin[d];
    if (l > 0) {
        const float* xr = x + ((size_t)b * L_SEQ + (l - 1)) * Z_DIM;
#pragma unroll
        for (int z = 0; z < Z_DIM; ++z) acc += xr[z] * Win[(size_t)z * D_MODEL + d];
    }
    int j = d >> 1;
    float arg = (float)l * __expf(-logf(10000.0f) * (2.0f * (float)j) / (float)D_MODEL);
    acc += (d & 1) ? __cosf(arg) : __sinf(arg);
    ctx[idx] = acc;
}

// ---------------------------------------------------------------------------
// VQ: for each (s,b) find argmin_k ||x[b,s]-cb[k]||^2, write cb[k] to memq[s,b]
// ---------------------------------------------------------------------------
__global__ __launch_bounds__(256) void vq_kernel(const float* __restrict__ x,
                                                 const float* __restrict__ cb,
                                                 float* __restrict__ memq)
{
    const int s = blockIdx.x, b = blockIdx.y;
    __shared__ float m[Z_DIM];
    __shared__ float bd[256];
    __shared__ int   bi[256];
    const int tid = threadIdx.x;
    if (tid < Z_DIM) m[tid] = x[((size_t)b * L_SEQ + s) * Z_DIM + tid];
    __syncthreads();
    float best = 3.4e38f; int bk = 0;
    for (int k = tid; k < K_CB; k += 256) {
        const float* c = cb + (size_t)k * Z_DIM;
        float d2 = 0.0f;
#pragma unroll
        for (int z = 0; z < Z_DIM; ++z) { float df = m[z] - c[z]; d2 += df * df; }
        if (d2 < best) { best = d2; bk = k; }
    }
    bd[tid] = best; bi[tid] = bk; __syncthreads();
    for (int st = 128; st > 0; st >>= 1) {
        if (tid < st) {
            if (bd[tid + st] < bd[tid] ||
                (bd[tid + st] == bd[tid] && bi[tid + st] < bi[tid])) {
                bd[tid] = bd[tid + st]; bi[tid] = bi[tid + st];
            }
        }
        __syncthreads();
    }
    int kbest = bi[0];
    if (tid < Z_DIM)
        memq[((size_t)s * B_SZ + b) * Z_DIM + tid] = cb[(size_t)kbest * Z_DIM + tid];
}

// mem_h16[r,d] = f16( memq[r,:] @ W_in + b_in )   (r = s*B+b, no PE)
__global__ void memh_kernel(const float* __restrict__ memq, const float* __restrict__ Win,
                            const float* __restrict__ bin, half_t* __restrict__ memh16)
{
    size_t idx = (size_t)blockIdx.x * blockDim.x + threadIdx.x;
    if (idx >= (size_t)L_SEQ * B_SZ * D_MODEL) return;
    int d = (int)(idx % D_MODEL);
    int r = (int)(idx / D_MODEL);
    const float* q = memq + (size_t)r * Z_DIM;
    float acc = bin[d];
#pragma unroll
    for (int z = 0; z < Z_DIM; ++z) acc += q[z] * Win[(size_t)z * D_MODEL + d];
    memh16[idx] = (half_t)acc;
}

// pack step-i working set: h rows (b,0..Lp-1) from ctx (b,0..Lp-1)
__global__ void pack_kernel(const float* __restrict__ ctx, float* __restrict__ h,
                            half_t* __restrict__ h16, int Lp)
{
    size_t idx = (size_t)blockIdx.x * blockDim.x + threadIdx.x;
    if (idx >= (size_t)B_SZ * Lp * D_MODEL) return;
    int d = (int)(idx % D_MODEL);
    int r = (int)(idx / D_MODEL);
    int l = r % Lp;
    int b = r / Lp;
    float v = ctx[((size_t)b * L_SEQ + l) * D_MODEL + d];
    h[idx] = v; h16[idx] = (half_t)v;
}

// ---------------------------------------------------------------------------
// Self-attention: scores + softmax (keys <= step) + AV, writes f16 attn output
// grid (Lp, B, H), block 64. qkv row layout: [q(512) | k(512) | v(512)]
// ---------------------------------------------------------------------------
__global__ __launch_bounds__(64) void attn_kernel(const float* __restrict__ qkv,
                                                  half_t* __restrict__ attn16,
                                                  int Lp, int step)
{
    const int q = blockIdx.x, b = blockIdx.y, hh = blockIdx.z;
    const int t = threadIdx.x;
    __shared__ float red[64];
    __shared__ float p[64];
    const float* qp = qkv + (size_t)(b * Lp + q) * (3 * D_MODEL) + hh * HDIM;
    float s = -1e30f;
    if (t <= step) {
        const float* kp = qkv + (size_t)(b * Lp + t) * (3 * D_MODEL) + D_MODEL + hh * HDIM;
        float acc = 0.0f;
#pragma unroll
        for (int d = 0; d < HDIM; ++d) acc += qp[d] * kp[d];
        s = acc * 0.125f;  // 1/sqrt(64)
    }
    red[t] = s; __syncthreads();
    for (int st = 32; st > 0; st >>= 1) { if (t < st) red[t] = fmaxf(red[t], red[t + st]); __syncthreads(); }
    float mx = red[0]; __syncthreads();
    float e = (t <= step) ? __expf(s - mx) : 0.0f;
    red[t] = e; __syncthreads();
    for (int st = 32; st > 0; st >>= 1) { if (t < st) red[t] += red[t + st]; __syncthreads(); }
    float denom = red[0];
    p[t] = e / denom; __syncthreads();
    float acc = 0.0f;
    const int kn = step + 1;
    for (int k = 0; k < kn; ++k)
        acc += p[k] * qkv[(size_t)(b * Lp + k) * (3 * D_MODEL) + 2 * D_MODEL + hh * HDIM + t];
    attn16[(size_t)(b * Lp + q) * D_MODEL + hh * HDIM + t] = (half_t)acc;
}

// ---------------------------------------------------------------------------
// h = LN1(h + o); optionally then h = LN2(h + ca[b]); writes h f32 + f16.
// One block (128 threads) per row.
// ---------------------------------------------------------------------------
__global__ __launch_bounds__(128) void ln_residual_kernel(
    float* __restrict__ h, const float* __restrict__ o,
    const float* __restrict__ ca,
    const float* __restrict__ g1, const float* __restrict__ b1,
    const float* __restrict__ g2, const float* __restrict__ b2,
    half_t* __restrict__ h16, int Lp)
{
    __shared__ float red[128];
    const int row = blockIdx.x;
    const int tid = threadIdx.x;
    const int b = row / Lp;
    float v[4];
    float s = 0.0f, sq = 0.0f;
#pragma unroll
    for (int j = 0; j < 4; ++j) {
        int d = tid + j * 128;
        float x = h[(size_t)row * D_MODEL + d] + o[(size_t)row * D_MODEL + d];
        v[j] = x; s += x; sq += x * x;
    }
    red[tid] = s; __syncthreads();
    for (int st = 64; st > 0; st >>= 1) { if (tid < st) red[tid] += red[tid + st]; __syncthreads(); }
    float mean = red[0] / D_MODEL; __syncthreads();
    red[tid] = sq; __syncthreads();
    for (int st = 64; st > 0; st >>= 1) { if (tid < st) red[tid] += red[tid + st]; __syncthreads(); }
    float var = red[0] / D_MODEL - mean * mean; __syncthreads();
    float inv = rsqrtf(var + 1e-5f);
#pragma unroll
    for (int j = 0; j < 4; ++j) {
        int d = tid + j * 128;
        v[j] = (v[j] - mean) * inv * g1[d] + b1[d];
    }
    if (ca) {
        s = 0.0f; sq = 0.0f;
#pragma unroll
        for (int j = 0; j < 4; ++j) {
            int d = tid + j * 128;
            float x = v[j] + ca[(size_t)b * D_MODEL + d];
            v[j] = x; s += x; sq += x * x;
        }
        red[tid] = s; __syncthreads();
        for (int st = 64; st > 0; st >>= 1) { if (tid < st) red[tid] += red[tid + st]; __syncthreads(); }
        float mean2 = red[0] / D_MODEL; __syncthreads();
        red[tid] = sq; __syncthreads();
        for (int st = 64; st > 0; st >>= 1) { if (tid < st) red[tid] += red[tid + st]; __syncthreads(); }
        float var2 = red[0] / D_MODEL - mean2 * mean2; __syncthreads();
        float inv2 = rsqrtf(var2 + 1e-5f);
#pragma unroll
        for (int j = 0; j < 4; ++j) {
            int d = tid + j * 128;
            v[j] = (v[j] - mean2) * inv2 * g2[d] + b2[d];
        }
    }
#pragma unroll
    for (int j = 0; j < 4; ++j) {
        int d = tid + j * 128;
        h[(size_t)row * D_MODEL + d] = v[j];
        h16[(size_t)row * D_MODEL + d] = (half_t)v[j];
    }
}

// ---------------------------------------------------------------------------
// Head: out[b,step,:] = LN(h[b,step,:]) @ W_out + b_out. One block per b.
// ---------------------------------------------------------------------------
__global__ __launch_bounds__(128) void head_kernel(
    const float* __restrict__ h, const float* __restrict__ g, const float* __restrict__ bb,
    const float* __restrict__ Wout, const float* __restrict__ bout,
    float* __restrict__ out, int Lp, int step)
{
    __shared__ float y[D_MODEL];
    __shared__ float red[128];
    const int b = blockIdx.x, tid = threadIdx.x;
    const float* hr = h + ((size_t)b * Lp + step) * D_MODEL;
    float v[4]; float s = 0.0f, sq = 0.0f;
#pragma unroll
    for (int j = 0; j < 4; ++j) { int d = tid + j * 128; v[j] = hr[d]; s += v[j]; sq += v[j] * v[j]; }
    red[tid] = s; __syncthreads();
    for (int st = 64; st > 0; st >>= 1) { if (tid < st) red[tid] += red[tid + st]; __syncthreads(); }
    float mean = red[0] / D_MODEL; __syncthreads();
    red[tid] = sq; __syncthreads();
    for (int st = 64; st > 0; st >>= 1) { if (tid < st) red[tid] += red[tid + st]; __syncthreads(); }
    float var = red[0] / D_MODEL - mean * mean; __syncthreads();
    float inv = rsqrtf(var + 1e-5f);
#pragma unroll
    for (int j = 0; j < 4; ++j) { int d = tid + j * 128; y[d] = (v[j] - mean) * inv * g[d] + bb[d]; }
    __syncthreads();
    if (tid < Z_DIM) {
        float acc = bout[tid];
        for (int d = 0; d < D_MODEL; ++d) acc += y[d] * Wout[(size_t)d * Z_DIM + tid];
        out[((size_t)b * L_SEQ + step) * Z_DIM + tid] = acc;
    }
}

// ---------------------------------------------------------------------------
// Host orchestration
// ---------------------------------------------------------------------------
static inline size_t alignup(size_t x) { return (x + 255) & ~(size_t)255; }

extern "C" void kernel_launch(void* const* d_in, const int* in_sizes, int n_in,
                              void* d_out, int out_size, void* d_ws, size_t ws_size,
                              hipStream_t stream)
{
    (void)in_sizes; (void)n_in; (void)out_size; (void)ws_size;
    const float* x        = (const float*)d_in[0];
    const float* codebook = (const float*)d_in[1];
    const float* W_in     = (const float*)d_in[2];
    const float* b_in     = (const float*)d_in[3];
    const float* sa_qkv_w = (const float*)d_in[4];
    const float* sa_qkv_b = (const float*)d_in[5];
    const float* sa_out_w = (const float*)d_in[6];
    const float* sa_out_b = (const float*)d_in[7];
    const float* ca_qkv_w = (const float*)d_in[8];
    const float* ca_qkv_b = (const float*)d_in[9];
    const float* ca_out_w = (const float*)d_in[10];
    const float* ca_out_b = (const float*)d_in[11];
    const float* ff1_w    = (const float*)d_in[12];
    const float* ff1_b    = (const float*)d_in[13];
    const float* ff2_w    = (const float*)d_in[14];
    const float* ff2_b    = (const float*)d_in[15];
    const float* ln1_g    = (const float*)d_in[16];
    const float* ln1_b    = (const float*)d_in[17];
    const float* ln2_g    = (const float*)d_in[18];
    const float* ln2_b    = (const float*)d_in[19];
    const float* ln3_g    = (const float*)d_in[20];
    const float* ln3_b    = (const float*)d_in[21];
    const float* ln_g     = (const float*)d_in[22];
    const float* ln_b     = (const float*)d_in[23];
    const float* W_out    = (const float*)d_in[24];
    const float* b_out    = (const float*)d_in[25];
    float* out = (float*)d_out;

    char* ws = (char*)d_ws;
    size_t off = 0;
    auto take = [&](size_t bytes) { char* p = ws + off; off += alignup(bytes); return p; };

    half_t* saqkvT  = (half_t*)take((size_t)NLAYER * 3 * D_MODEL * D_MODEL * 2);
    half_t* saoutT  = (half_t*)take((size_t)NLAYER * D_MODEL * D_MODEL * 2);
    half_t* cavT    = (half_t*)take((size_t)NLAYER * D_MODEL * D_MODEL * 2);
    half_t* caoutT  = (half_t*)take((size_t)NLAYER * D_MODEL * D_MODEL * 2);
    half_t* ff1T    = (half_t*)take((size_t)NLAYER * DFF_ * D_MODEL * 2);
    half_t* ff2T    = (half_t*)take((size_t)NLAYER * D_MODEL * DFF_ * 2);
    float*  ctx0    = (float*) take((size_t)B_SZ * L_SEQ * D_MODEL * 4);
    float*  memq    = (float*) take((size_t)L_SEQ * B_SZ * Z_DIM * 4);
    half_t* memh16  = (half_t*)take((size_t)L_SEQ * B_SZ * D_MODEL * 2);
    half_t* cav16   = (half_t*)take((size_t)L_SEQ * B_SZ * D_MODEL * 2);
    float*  ca_add  = (float*) take((size_t)NLAYER * L_SEQ * B_SZ * D_MODEL * 4);
    float*  h       = (float*) take((size_t)B_SZ * L_SEQ * D_MODEL * 4);
    half_t* h16     = (half_t*)take((size_t)B_SZ * L_SEQ * D_MODEL * 2);
    float*  qkv     = (float*) take((size_t)B_SZ * L_SEQ * 3 * D_MODEL * 4);
    half_t* attn16  = (half_t*)take((size_t)B_SZ * L_SEQ * D_MODEL * 2);
    float*  tmp     = (float*) take((size_t)B_SZ * L_SEQ * D_MODEL * 4);
    half_t* ff16    = (half_t*)take((size_t)B_SZ * L_SEQ * DFF_ * 2);

    auto cdiv = [](size_t a, size_t b) { return (unsigned)((a + b - 1) / b); };

    // ---- one-time prep: weight convert/transpose to f16 ----
    {
        size_t n;
        n = (size_t)NLAYER * (3 * D_MODEL) * D_MODEL;
        wT_kernel<<<cdiv(n, 256), 256, 0, stream>>>(sa_qkv_w, saqkvT, NLAYER, D_MODEL, 3 * D_MODEL, 0, 3 * D_MODEL);
        n = (size_t)NLAYER * D_MODEL * D_MODEL;
        wT_kernel<<<cdiv(n, 256), 256, 0, stream>>>(sa_out_w, saoutT, NLAYER, D_MODEL, D_MODEL, 0, D_MODEL);
        wT_kernel<<<cdiv(n, 256), 256, 0, stream>>>(ca_qkv_w, cavT,   NLAYER, D_MODEL, 3 * D_MODEL, 2 * D_MODEL, D_MODEL);
        wT_kernel<<<cdiv(n, 256), 256, 0, stream>>>(ca_out_w, caoutT, NLAYER, D_MODEL, D_MODEL, 0, D_MODEL);
        n = (size_t)NLAYER * DFF_ * D_MODEL;
        wT_kernel<<<cdiv(n, 256), 256, 0, stream>>>(ff1_w, ff1T, NLAYER, D_MODEL, DFF_, 0, DFF_);
        wT_kernel<<<cdiv(n, 256), 256, 0, stream>>>(ff2_w, ff2T, NLAYER, DFF_, D_MODEL, 0, D_MODEL);
    }
    // context embedding (shared across all steps)
    ctx_kernel<<<cdiv((size_t)B_SZ * L_SEQ * D_MODEL, 256), 256, 0, stream>>>(x, W_in, b_in, ctx0);
    // VQ + memory embedding for all 64 steps
    vq_kernel<<<dim3(L_SEQ, B_SZ), 256, 0, stream>>>(x, codebook, memq);
    memh_kernel<<<cdiv((size_t)L_SEQ * B_SZ * D_MODEL, 256), 256, 0, stream>>>(memq, W_in, b_in, memh16);
    // Cross-attention degenerates to broadcast of out_proj(V_proj(mem)); precompute per layer.
    for (int l = 0; l < NLAYER; ++l) {
        const int M = L_SEQ * B_SZ;  // 2048
        gemm_wmma_kernel<<<dim3(D_MODEL / 64, M / 32), 32, 0, stream>>>(
            memh16, cavT + (size_t)l * D_MODEL * D_MODEL,
            ca_qkv_b + (size_t)l * 3 * D_MODEL + 2 * D_MODEL,
            nullptr, cav16, M, D_MODEL, D_MODEL, 0);
        gemm_wmma_kernel<<<dim3(D_MODEL / 64, M / 32), 32, 0, stream>>>(
            cav16, caoutT + (size_t)l * D_MODEL * D_MODEL,
            ca_out_b + (size_t)l * D_MODEL,
            ca_add + (size_t)l * L_SEQ * B_SZ * D_MODEL, nullptr, M, D_MODEL, D_MODEL, 0);
    }

    // ---- 64 decode steps; step i only needs rows 0..i (causal truncation) ----
    for (int i = 0; i < L_SEQ; ++i) {
        const int Lp = ((i + 1) + 15) & ~15;  // padded to tile multiple
        const int M = B_SZ * Lp;              // multiple of 512
        pack_kernel<<<cdiv((size_t)M * D_MODEL, 256), 256, 0, stream>>>(ctx0, h, h16, Lp);
        for (int l = 0; l < NLAYER; ++l) {
            // self-attention
            gemm_wmma_kernel<<<dim3(3 * D_MODEL / 64, M / 32), 32, 0, stream>>>(
                h16, saqkvT + (size_t)l * 3 * D_MODEL * D_MODEL,
                sa_qkv_b + (size_t)l * 3 * D_MODEL, qkv, nullptr, M, 3 * D_MODEL, D_MODEL, 0);
            attn_kernel<<<dim3(Lp, B_SZ, NHEAD), 64, 0, stream>>>(qkv, attn16, Lp, i);
            gemm_wmma_kernel<<<dim3(D_MODEL / 64, M / 32), 32, 0, stream>>>(
                attn16, saoutT + (size_t)l * D_MODEL * D_MODEL,
                sa_out_b + (size_t)l * D_MODEL, tmp, nullptr, M, D_MODEL, D_MODEL, 0);
            // LN1 then fused cross-attention add + LN2
            ln_residual_kernel<<<M, 128, 0, stream>>>(
                h, tmp, ca_add + ((size_t)l * L_SEQ + i) * B_SZ * D_MODEL,
                ln1_g + (size_t)l * D_MODEL, ln1_b + (size_t)l * D_MODEL,
                ln2_g + (size_t)l * D_MODEL, ln2_b + (size_t)l * D_MODEL, h16, Lp);
            // FFN
            gemm_wmma_kernel<<<dim3(DFF_ / 64, M / 32), 32, 0, stream>>>(
                h16, ff1T + (size_t)l * DFF_ * D_MODEL,
                ff1_b + (size_t)l * DFF_, nullptr, ff16, M, DFF_, D_MODEL, 1);
            gemm_wmma_kernel<<<dim3(D_MODEL / 64, M / 32), 32, 0, stream>>>(
                ff16, ff2T + (size_t)l * D_MODEL * DFF_,
                ff2_b + (size_t)l * D_MODEL, tmp, nullptr, M, D_MODEL, DFF_, 0);
            ln_residual_kernel<<<M, 128, 0, stream>>>(
                h, tmp, nullptr,
                ln3_g + (size_t)l * D_MODEL, ln3_b + (size_t)l * D_MODEL,
                nullptr, nullptr, h16, Lp);
        }
        head_kernel<<<B_SZ, 128, 0, stream>>>(h, ln_g, ln_b, W_out, b_out, out, Lp, i);
    }
}
